// vgg16_19524921328218
// MI455X (gfx1250) — matrix operand order repaired and checked by reading, not compiled, not run on hardware
//
#include <hip/hip_runtime.h>
#include <hip/hip_bf16.h>

typedef __attribute__((ext_vector_type(16))) __bf16    v16bf;
typedef __attribute__((ext_vector_type(8)))  float     v8f;
typedef __attribute__((ext_vector_type(8)))  unsigned  v8u;
typedef __attribute__((ext_vector_type(4)))  unsigned  v4u;

// ---------------------------------------------------------------------------
// Weight repack: fp32 [Cout, 7*Cin] -> bf16 [Cout, Kpad], channel-padded so
// that j = k >> cshift, c = k & (CinPad-1).  Out-of-range (j>=7 or c>=Cin) = 0.
// ---------------------------------------------------------------------------
__global__ void wpack_kernel(const float* __restrict__ W, __bf16* __restrict__ Wb,
                             int Cout, int Cin, int cshift, int Kpad) {
    int i = blockIdx.x * 256 + threadIdx.x;
    if (i >= Cout * Kpad) return;
    int o  = i / Kpad;
    int kp = i - o * Kpad;
    int j  = kp >> cshift;
    int c  = kp & ((1 << cshift) - 1);
    float val = (j < 7 && c < Cin) ? W[(size_t)o * 7 * Cin + j * Cin + c] : 0.f;
    Wb[i] = (__bf16)val;
}

// x fp32 [N,3] -> bf16 [N,4] (channel 3 zero-padded)
__global__ void xpad_kernel(const float* __restrict__ x, __bf16* __restrict__ out, int N) {
    int i = blockIdx.x * 256 + threadIdx.x;
    if (i >= N * 4) return;
    int v = i >> 2, c = i & 3;
    out[i] = (__bf16)((c < 3) ? x[v * 3 + c] : 0.f);
}

// ---------------------------------------------------------------------------
// Layer-0 conv: CinPad=4, Kpad=32, Kact=28 -> exactly one WMMA per tile.
// Fragment mappings (A and B both in contiguous k order):
//   A 16x32: lane l -> M=l&15, kbase=16*(l>>4); pair p: k=kbase+2*p
//   B 32x16: lane l -> col l&15, K=16*(l>>4)+e  (32 contiguous bytes of W row)
//   D 16x16: VGPR r -> row r+8*(l>>4), col l&15
// ---------------------------------------------------------------------------
__global__ void dine_conv_wmma_l0(const __bf16* __restrict__ act,
                                  const int*    __restrict__ neigh,
                                  const __bf16* __restrict__ Wb,
                                  const float*  __restrict__ bias,
                                  float*        __restrict__ out,
                                  int N, int Cout) {
    const int lane = threadIdx.x & 31;
    const int wave = threadIdx.x >> 5;
    const int mt   = blockIdx.x * 8 + wave;
    if (mt * 16 >= N) return;

    const int half = lane >> 4;
    const int m16  = lane & 15;
    int v = mt * 16 + m16;
    if (v >= N) v = N - 1;
    const int* nrow = neigh + v * 7;
    const int col = blockIdx.y * 16 + m16;

    v16bf bfrag = *(const v16bf*)(Wb + (size_t)col * 32 + half * 16);
    v8u au;
    const int kb = half * 16;
#pragma unroll
    for (int p = 0; p < 8; ++p) {
        int k = kb + p * 2;
        unsigned pv = 0u;
        if (k < 28) {                       // Kact = 28; beyond is zero padding
            int j = k >> 2;
            int c = k & 3;
            int src = nrow[j];
            pv = *(const unsigned*)(act + ((size_t)src << 2) + c);
        }
        au[p] = pv;
    }
    v16bf afrag = __builtin_bit_cast(v16bf, au);
    v8f acc = {};
    acc = __builtin_amdgcn_wmma_f32_16x16x32_bf16(
              false, afrag, false, bfrag, (short)0, acc, false, false);

    const float bv = bias[col];
#pragma unroll
    for (int r = 0; r < 8; ++r) {
        int row = mt * 16 + r + half * 8;
        if (row < N) out[(size_t)row * Cout + col] = acc[r] + bv;
    }
}

// ---------------------------------------------------------------------------
// Main conv (Cin >= 32): gather + GEMM. Weight tile double-buffered in LDS via
// async global->LDS copies (ASYNCcnt), shared by all 8 waves of the block.
// Block = 256 threads = 8 waves; block computes 256 rows x 16 out-channels
// (2 M-tiles per wave: one LDS B-fragment feeds two WMMAs -> two independent
// accumulator chains, 16x reuse of each staged chunk).
// K chunk = 128: full chunks fully unrolled (8 static WMMAs), tails rolled.
// ---------------------------------------------------------------------------
#define KCHUNK      128
#define BROW_STRIDE 136   // bf16 elements per LDS row: 128 + 8 pad (bank spread)

__global__ void dine_conv_wmma_lds(const __bf16* __restrict__ act,
                                   const int*    __restrict__ neigh,
                                   const __bf16* __restrict__ Wb,
                                   const float*  __restrict__ bias,
                                   float*        __restrict__ out,
                                   int N, int Cout, int Kpad, int cshift) {
    __shared__ __bf16 Bsm[2][16 * BROW_STRIDE];

    const int tid  = threadIdx.x;
    const int lane = tid & 31;
    const int wave = tid >> 5;
    const int half = lane >> 4;
    const int m16  = lane & 15;

    const int rbase = blockIdx.x * 256 + wave * 32;   // 2 M-tiles per wave
    int va = rbase + m16;          if (va >= N) va = N - 1;
    int vb = rbase + 16 + m16;     if (vb >= N) vb = N - 1;
    const int* nrowA = neigh + va * 7;
    const int* nrowB = neigh + vb * 7;

    const int col  = blockIdx.y * 16 + m16;
    const int col0 = blockIdx.y * 16;
    const int cmask = (1 << cshift) - 1;
    const int nchunks = (Kpad + KCHUNK - 1) / KCHUNK;

    // Stage one K-chunk of the 16-column weight tile into LDS buffer `buf`.
    // One b128 async load per participating thread.
    auto stage = [&](int k0, int buf) {
        int kc = Kpad - k0; if (kc > KCHUNK) kc = KCHUNK;
        int slots = kc >> 3;              // 16B slots per column
        int total = slots << 4;           // 16 columns
        if (tid < total) {
            int c   = tid / slots;
            int seg = tid - c * slots;
            const __bf16* gp = Wb + (size_t)(col0 + c) * Kpad + k0 + seg * 8;
            unsigned lo = (unsigned)(size_t)&Bsm[buf][c * BROW_STRIDE + seg * 8];
            unsigned long long ga = (unsigned long long)(size_t)gp;
            asm volatile("global_load_async_to_lds_b128 %0, %1, off"
                         :: "v"(lo), "v"(ga) : "memory");
        }
    };

    v8f acc0 = {};
    v8f acc1 = {};

    // One K-step (32 wide): LDS B-fragment + two gathered A-fragments + 2 WMMAs.
    auto kstep = [&](int k0, const __bf16* bp) {
        v16bf bfrag = *(const v16bf*)bp;              // 32B from LDS
        const int j  = k0 >> cshift;                  // one neighbor slot per step
        const int co = (k0 & cmask) + half * 16;
        const int sA = nrowA[j];
        const int sB = nrowB[j];
        const __bf16* aA = act + ((size_t)sA << cshift) + co;
        const __bf16* aB = act + ((size_t)sB << cshift) + co;
        v4u la = *(const v4u*)aA; v4u ha = *(const v4u*)(aA + 8);
        v4u lb = *(const v4u*)aB; v4u hb = *(const v4u*)(aB + 8);
        v8u ua = {la.x, la.y, la.z, la.w, ha.x, ha.y, ha.z, ha.w};
        v8u ub = {lb.x, lb.y, lb.z, lb.w, hb.x, hb.y, hb.z, hb.w};
        acc0 = __builtin_amdgcn_wmma_f32_16x16x32_bf16(
                   false, __builtin_bit_cast(v16bf, ua), false, bfrag,
                   (short)0, acc0, false, false);
        acc1 = __builtin_amdgcn_wmma_f32_16x16x32_bf16(
                   false, __builtin_bit_cast(v16bf, ub), false, bfrag,
                   (short)0, acc1, false, false);
    };

    stage(0, 0);
    for (int ci = 0; ci < nchunks; ++ci) {
        asm volatile("s_wait_asynccnt 0x0" ::: "memory");
        __syncthreads();                              // staged chunk visible to all
        if (ci + 1 < nchunks) stage((ci + 1) * KCHUNK, (ci + 1) & 1);

        const int k0c = ci * KCHUNK;
        int kc = Kpad - k0c; if (kc > KCHUNK) kc = KCHUNK;
        const __bf16* bsm = &Bsm[ci & 1][m16 * BROW_STRIDE + half * 16];

        if (kc == KCHUNK) {
#pragma unroll
            for (int u = 0; u < KCHUNK / 32; ++u)
                kstep(k0c + u * 32, bsm + u * 32);
        } else {
            for (int kk = 0; kk < kc; kk += 32)
                kstep(k0c + kk, bsm + kk);
        }
        __syncthreads();                              // reads done before buffer reuse
    }

    const float bv = bias[col];
#pragma unroll
    for (int r = 0; r < 8; ++r) {
        int rowA = rbase + r + half * 8;
        int rowB = rowA + 16;
        if (rowA < N) out[(size_t)rowA * Cout + col] = acc0[r] + bv;
        if (rowB < N) out[(size_t)rowB * Cout + col] = acc1[r] + bv;
    }
}

// ---------------------------------------------------------------------------
// BatchNorm statistics over vertices: one block per channel.
// stats[c] = mean, stats[C+c] = rsqrt(var + eps)
// ---------------------------------------------------------------------------
__global__ void bn_stats_kernel(const float* __restrict__ y, float* __restrict__ stats,
                                int N, int C) {
    __shared__ float ssum[256];
    __shared__ float ssq[256];
    const int c = blockIdx.x;
    float s = 0.f, q = 0.f;
    for (int v = threadIdx.x; v < N; v += 256) {
        float x = y[(size_t)v * C + c];
        s += x; q += x * x;
    }
    ssum[threadIdx.x] = s; ssq[threadIdx.x] = q;
    __syncthreads();
    for (int w = 128; w > 0; w >>= 1) {
        if (threadIdx.x < w) {
            ssum[threadIdx.x] += ssum[threadIdx.x + w];
            ssq[threadIdx.x]  += ssq[threadIdx.x + w];
        }
        __syncthreads();
    }
    if (threadIdx.x == 0) {
        float m   = ssum[0] / (float)N;
        float var = ssq[0] / (float)N - m * m;
        stats[c]     = m;
        stats[C + c] = rsqrtf(var + 1e-5f);
    }
}

// BN apply + LeakyReLU(0.1) + cast to bf16 (input for the next gather-GEMM)
__global__ void bn_apply_kernel(const float* __restrict__ y, const float* __restrict__ stats,
                                const float* __restrict__ g, const float* __restrict__ b,
                                __bf16* __restrict__ out, int total, int cmask, int C) {
    int i = blockIdx.x * 256 + threadIdx.x;
    if (i >= total) return;
    int c = i & cmask;
    float val = (y[i] - stats[c]) * stats[C + c] * g[c] + b[c];
    val = (val >= 0.f) ? val : 0.1f * val;
    out[i] = (__bf16)val;
}

// Mean-pool to coarser level: out[u,c] = mean_j act[neigh[u*7+j], c]
__global__ void pool_kernel(const __bf16* __restrict__ act, const int* __restrict__ neigh,
                            __bf16* __restrict__ out, int n, int cshift) {
    int i = blockIdx.x * 256 + threadIdx.x;
    if (i >= (n << cshift)) return;
    int u = i >> cshift;
    int c = i & ((1 << cshift) - 1);
    const int* nr = neigh + u * 7;
    float s = 0.f;
#pragma unroll
    for (int j = 0; j < 7; ++j)
        s += (float)act[((size_t)nr[j] << cshift) + c];
    out[i] = (__bf16)(s * (1.f / 7.f));
}

// Global mean over 162 vertices + 2x1024 FC. Single block.
__global__ void head_kernel(const __bf16* __restrict__ act,
                            const float* __restrict__ fcw, const float* __restrict__ fcb,
                            float* __restrict__ out) {
    __shared__ float mean[1024];
    for (int c = threadIdx.x; c < 1024; c += 256) {
        float s = 0.f;
        for (int v = 0; v < 162; ++v) s += (float)act[(size_t)v * 1024 + c];
        mean[c] = s * (1.f / 162.f);
    }
    __syncthreads();
    if (threadIdx.x < 2) {
        float s = fcb[threadIdx.x];
        for (int c = 0; c < 1024; ++c) s += fcw[threadIdx.x * 1024 + c] * mean[c];
        out[threadIdx.x] = s;
    }
}

// ---------------------------------------------------------------------------
// Host orchestration
// ---------------------------------------------------------------------------
static inline int ceil_div(int a, int b) { return (a + b - 1) / b; }

extern "C" void kernel_launch(void* const* d_in, const int* in_sizes, int n_in,
                              void* d_out, int out_size, void* d_ws, size_t ws_size,
                              hipStream_t stream) {
    (void)in_sizes; (void)n_in; (void)out_size; (void)ws_size;

    // Input layout (setup_inputs dict order, lists flattened element-wise):
    // 0:x  1..6:n0..n5  7..19:conv_ws  20..32:conv_bs  33..45:bn_gs  46..58:bn_bs
    // 59:fc_w  60:fc_b
    const float* x = (const float*)d_in[0];
    const int* neighs[6];
    for (int i = 0; i < 6; ++i) neighs[i] = (const int*)d_in[1 + i];
    const float *Wf[13], *Bf[13], *Gf[13], *Bb[13];
    for (int i = 0; i < 13; ++i) {
        Wf[i] = (const float*)d_in[7  + i];
        Bf[i] = (const float*)d_in[20 + i];
        Gf[i] = (const float*)d_in[33 + i];
        Bb[i] = (const float*)d_in[46 + i];
    }
    const float* fcw = (const float*)d_in[59];
    const float* fcb = (const float*)d_in[60];

    static const int NS[6]     = {163842, 40962, 10242, 2562, 642, 162};
    static const int LVL[13]   = {0,0,0,1,1,2,2,3,3,4,4,5,5};
    static const int POOL[13]  = {0,0,0,1,0,1,0,1,0,1,0,1,0};
    static const int CIN[13]   = {3,32,32,32,64,64,128,128,256,256,512,512,1024};
    static const int COUT[13]  = {32,32,32,64,64,128,128,256,256,512,512,1024,1024};
    static const int CSH[13]   = {2,5,5,5,6,6,7,7,8,8,9,9,10};          // log2(CinPad)
    static const int KPAD[13]  = {32,224,224,224,448,448,896,896,1792,1792,3584,3584,7168};
    static const int COSH[13]  = {5,5,5,6,6,7,7,8,8,9,9,10,10};         // log2(Cout)

    // Workspace carve-up
    char* ws = (char*)d_ws;
    size_t off = 0;
    auto carve = [&](size_t bytes) -> char* {
        char* p = ws + off;
        off += (bytes + 255) & ~(size_t)255;
        return p;
    };
    __bf16* Wb[13];
    for (int i = 0; i < 13; ++i)
        Wb[i] = (__bf16*)carve((size_t)COUT[i] * KPAD[i] * sizeof(__bf16));
    __bf16* actA = (__bf16*)carve((size_t)163842 * 32 * sizeof(__bf16));
    __bf16* actB = (__bf16*)carve((size_t)163842 * 32 * sizeof(__bf16));
    float*  conv = (float*) carve((size_t)163842 * 32 * sizeof(float));
    float*  stats = (float*)carve((size_t)2 * 1024 * sizeof(float));

    // 1) pack weights to bf16 (padded layout), pad input x to [N,4] bf16
    for (int i = 0; i < 13; ++i) {
        int total = COUT[i] * KPAD[i];
        wpack_kernel<<<ceil_div(total, 256), 256, 0, stream>>>(
            Wf[i], Wb[i], COUT[i], CIN[i], CSH[i], KPAD[i]);
    }
    xpad_kernel<<<ceil_div(163842 * 4, 256), 256, 0, stream>>>(x, actA, 163842);

    // 2) layer chain
    __bf16* cur = actA;
    __bf16* alt = actB;
    for (int i = 0; i < 13; ++i) {
        const int lvl = LVL[i];
        const int N   = NS[lvl];
        if (POOL[i]) {
            int total = N << CSH[i];
            pool_kernel<<<ceil_div(total, 256), 256, 0, stream>>>(
                cur, neighs[lvl - 1], alt, N, CSH[i]);
            __bf16* t = cur; cur = alt; alt = t;
        }
        if (i == 0) {
            dim3 grid(ceil_div(N, 128), COUT[i] / 16);
            dine_conv_wmma_l0<<<grid, 256, 0, stream>>>(
                cur, neighs[lvl], Wb[i], Bf[i], conv, N, COUT[i]);
        } else {
            dim3 grid(ceil_div(N, 256), COUT[i] / 16);
            dine_conv_wmma_lds<<<grid, 256, 0, stream>>>(
                cur, neighs[lvl], Wb[i], Bf[i], conv, N, COUT[i], KPAD[i], CSH[i]);
        }

        bn_stats_kernel<<<COUT[i], 256, 0, stream>>>(conv, stats, N, COUT[i]);
        int total = N * COUT[i];
        bn_apply_kernel<<<ceil_div(total, 256), 256, 0, stream>>>(
            conv, stats, Gf[i], Bb[i], alt, total, (1 << COSH[i]) - 1, COUT[i]);
        __bf16* t = cur; cur = alt; alt = t;
    }

    // 3) global mean + FC -> [1,2]
    head_kernel<<<1, 256, 0, stream>>>(cur, fcw, fcb, (float*)d_out);
}
